// EdgeFeat_41412074668753
// MI455X (gfx1250) — matrix operand order repaired and checked by reading, not compiled, not run on hardware
//
#include <hip/hip_runtime.h>

// ---------------------------------------------------------------------------
// EdgeFeat fused kernel for MI455X (gfx1250, wave32, WMMA)
//   e[r,c]   = n[i,c] * n[j,c] * s[r,c]        (r = i*512 + j)
//   y[r,d]   = sum_c e[r,c] * W[d,c] + b[d]    (bf16 hi/lo split WMMA GEMM)
//   out[r,d] = y[r,d] * v[r,d]
// Memory-bound problem (~768 MB stream @ 23.3 TB/s); WMMA work is cheap, so
// we spend 2x WMMA (bf16 split of e) to keep near-fp32 accuracy.
// ---------------------------------------------------------------------------

typedef __attribute__((ext_vector_type(4)))  float   f4;
typedef __attribute__((ext_vector_type(8)))  float   v8f;
typedef __attribute__((ext_vector_type(8)))  __bf16  v8bf;
typedef __attribute__((ext_vector_type(16))) __bf16  v16bf;

#define NN      512                 // nodes
#define CD      256                 // channels
#define RB_ROWS 16                  // edge rows per tile (WMMA M)
#define NUM_RB  ((NN * NN) / RB_ROWS)  // 16384 row-blocks
#define WST     264                 // LDS row stride in halfs (256 + 8 pad)
#define GRIDX   1024
#define BLOCKX  256                 // 8 waves of 32

static __device__ __forceinline__ v16bf cat16(v8bf a, v8bf b) {
  return __builtin_shufflevector(a, b, 0,1,2,3,4,5,6,7,8,9,10,11,12,13,14,15);
}

__global__ __launch_bounds__(BLOCKX)
void EdgeFeat_41412074668753_kernel(const float* __restrict__ nmat,   // (512,256)
                                    const float* __restrict__ smat,   // (512,512,256)
                                    const float* __restrict__ vmat,   // (512,512,256)
                                    const float* __restrict__ Wmat,   // (256,256)
                                    const float* __restrict__ bvec,   // (256,)
                                    float* __restrict__ out)          // (512,512,256)
{
  // LDS: W in bf16 (132KB, padded rows) + e tile hi/lo (2 x 8.25KB) ~= 152KB
  __shared__ __bf16 Wlds[CD * WST];
  __shared__ __bf16 Ehi[RB_ROWS * WST];
  __shared__ __bf16 Elo[RB_ROWS * WST];

  const int tid  = threadIdx.x;
  const int lane = tid & 31;
  const int wave = tid >> 5;

  // ---- one-time: W (f32 row-major [d][c]) -> bf16 in LDS ----------------
  {
    const float* wrow = Wmat + (size_t)tid * CD;   // thread t owns W row t
    __bf16* dst = &Wlds[tid * WST];
    #pragma unroll 4
    for (int c = 0; c < CD; c += 8) {
      f4 w0 = *(const f4*)(wrow + c);
      f4 w1 = *(const f4*)(wrow + c + 4);
      v8bf h;
      h[0]=(__bf16)w0.x; h[1]=(__bf16)w0.y; h[2]=(__bf16)w0.z; h[3]=(__bf16)w0.w;
      h[4]=(__bf16)w1.x; h[5]=(__bf16)w1.y; h[6]=(__bf16)w1.z; h[7]=(__bf16)w1.w;
      *(v8bf*)(dst + c) = h;                       // ds_store_b128
    }
  }
  __syncthreads();

  // staging role: thread t -> row (t>>4), 16 channels starting at (t&15)*16
  const int erow = tid >> 4;
  const int ec0  = (tid & 15) << 4;

  // WMMA fragment lane mapping (wave32):
  //  A (16x32 bf16): lane<16 -> halfs 0-7 = K+0..7, halfs 8-15 = K+16..23
  //                  lane>=16 same with +8 K offset  -> two b128 LDS reads
  //  B (32x16 bf16): lane holds column d = tile*16 + (lane&15),
  //                  halfs 0-7 = K+0..7, halfs 8-15 = K+16..23 (+8 for lanes>=16)
  const int m    = lane & 15;
  const int asel = (lane >> 4) << 3;   // 0 or 8  (halfs)
  const int bsel = (lane >> 4) << 3;   // 0 or 8  (halfs)
  const int d0   = (wave << 4) + m;          // d-tile = wave
  const int d1   = ((wave + 8) << 4) + m;    // d-tile = wave+8
  const float bias0 = bvec[d0];
  const float bias1 = bvec[d1];
  const int rowoff = (lane >> 4) << 3;  // C/D layout: lanes>=16 hold M+8

  for (int rb = blockIdx.x; rb < NUM_RB; rb += gridDim.x) {
    const int rbase = rb << 4;          // first edge row of tile
    const int i     = rb >> 5;          // 16 rows share node i (512/16=32)
    const int j0    = (rb & 31) << 4;   // first j of tile

    // ---- stage e = n_i * n_j * s  ->  bf16 hi/lo in LDS -----------------
    {
      const float* nip = nmat + (size_t)i * CD + ec0;
      const float* njp = nmat + (size_t)(j0 + erow) * CD + ec0;
      const float* sp  = smat + (size_t)(rbase + erow) * CD + ec0;
      if (rb + (int)gridDim.x < NUM_RB)  // gfx1250 global_prefetch_b8
        __builtin_prefetch(sp + (size_t)gridDim.x * RB_ROWS * CD, 0, 1);

      __bf16* hdst = &Ehi[erow * WST + ec0];
      __bf16* ldst = &Elo[erow * WST + ec0];
      #pragma unroll
      for (int u = 0; u < 16; u += 8) {
        f4 a0 = *(const f4*)(nip + u);
        f4 a1 = *(const f4*)(nip + u + 4);
        f4 g0 = *(const f4*)(njp + u);
        f4 g1 = *(const f4*)(njp + u + 4);
        f4 s0 = __builtin_nontemporal_load((const f4*)(sp + u));     // stream
        f4 s1 = __builtin_nontemporal_load((const f4*)(sp + u + 4));
        f4 e0 = a0 * g0 * s0;
        f4 e1 = a1 * g1 * s1;
        float ef[8] = {e0.x, e0.y, e0.z, e0.w, e1.x, e1.y, e1.z, e1.w};
        v8bf hv, lv;
        #pragma unroll
        for (int q = 0; q < 8; ++q) {
          __bf16 h = (__bf16)ef[q];
          hv[q] = h;
          lv[q] = (__bf16)(ef[q] - (float)h);  // residual for fp32-ish GEMM
        }
        *(v8bf*)(hdst + u) = hv;               // ds_store_b128
        *(v8bf*)(ldst + u) = lv;
      }
    }
    __syncthreads();

    // ---- GEMM: 16 rows x 32 d-cols per wave, K = 256 = 8 x 32 -----------
    v8f acc0 = {0.f,0.f,0.f,0.f,0.f,0.f,0.f,0.f};
    v8f acc1 = acc0;
    #pragma unroll
    for (int k = 0; k < 8; ++k) {
      const int kc = k << 5;  // K offset in halfs
      v16bf Ah = cat16(*(const v8bf*)&Ehi[m * WST + kc + asel],
                       *(const v8bf*)&Ehi[m * WST + kc + 16 + asel]);
      v16bf Al = cat16(*(const v8bf*)&Elo[m * WST + kc + asel],
                       *(const v8bf*)&Elo[m * WST + kc + 16 + asel]);
      v16bf B0 = cat16(*(const v8bf*)&Wlds[d0 * WST + kc + bsel],
                       *(const v8bf*)&Wlds[d0 * WST + kc + bsel + 16]);
      v16bf B1 = cat16(*(const v8bf*)&Wlds[d1 * WST + kc + bsel],
                       *(const v8bf*)&Wlds[d1 * WST + kc + bsel + 16]);
      acc0 = __builtin_amdgcn_wmma_f32_16x16x32_bf16(false, Ah, false, B0,
                                                     (short)0, acc0, false, false);
      acc0 = __builtin_amdgcn_wmma_f32_16x16x32_bf16(false, Al, false, B0,
                                                     (short)0, acc0, false, false);
      acc1 = __builtin_amdgcn_wmma_f32_16x16x32_bf16(false, Ah, false, B1,
                                                     (short)0, acc1, false, false);
      acc1 = __builtin_amdgcn_wmma_f32_16x16x32_bf16(false, Al, false, B1,
                                                     (short)0, acc1, false, false);
    }

    // ---- epilogue: out = (acc + b) * v, streamed non-temporal -----------
    #pragma unroll
    for (int p = 0; p < 8; ++p) {
      const size_t row = (size_t)(rbase + rowoff + p);
      const size_t i0  = row * CD + d0;
      const size_t i1  = row * CD + d1;
      float v0 = __builtin_nontemporal_load(vmat + i0);
      float v1 = __builtin_nontemporal_load(vmat + i1);
      __builtin_nontemporal_store((acc0[p] + bias0) * v0, out + i0);
      __builtin_nontemporal_store((acc1[p] + bias1) * v1, out + i1);
    }
    __syncthreads();  // protect Ehi/Elo before next iteration's staging
  }
}

extern "C" void kernel_launch(void* const* d_in, const int* in_sizes, int n_in,
                              void* d_out, int out_size, void* d_ws, size_t ws_size,
                              hipStream_t stream) {
  (void)in_sizes; (void)n_in; (void)out_size; (void)d_ws; (void)ws_size;
  const float* nmat = (const float*)d_in[0];
  const float* smat = (const float*)d_in[1];
  const float* vmat = (const float*)d_in[2];
  const float* Wmat = (const float*)d_in[3];
  const float* bvec = (const float*)d_in[4];
  float* out = (float*)d_out;
  EdgeFeat_41412074668753_kernel<<<GRIDX, BLOCKX, 0, stream>>>(
      nmat, smat, vmat, Wmat, bvec, out);
}
